// MultiheadAttentionAttrExtract_14731737825915
// MI455X (gfx1250) — matrix operand order repaired and checked by reading, not compiled, not run on hardware
//
#include <hip/hip_runtime.h>
#include <hip/hip_bf16.h>

// Shapes
#define S_ 1024
#define B_ 8
#define D_ 768
#define H_ 12
#define DH_ 64
#define TOKENS_ 1016            // S - PROMPT_NUM
#define QSZ_ (S_ * B_ * D_)     // 6291456 elements per output tensor

typedef __attribute__((ext_vector_type(16))) __bf16 v16bf;
typedef __attribute__((ext_vector_type(2)))  __bf16 v2bf;
typedef __attribute__((ext_vector_type(8)))  float  v8f;

union Frag16 {
    v16bf v;
    unsigned int d[8];
};

// Native bf16 conversion
__device__ __forceinline__ unsigned short f2bfu(float f) {
    union { __bf16 h; unsigned short u; } cv;
    cv.h = (__bf16)f;
    return cv.u;
}

#if __has_builtin(__builtin_amdgcn_cvt_pk_bf16_f32)
__device__ __forceinline__ unsigned int pack2bf(float lo, float hi) {
    union { v2bf v; unsigned int u; } cv;
    cv.v = __builtin_amdgcn_cvt_pk_bf16_f32(lo, hi);   // v_cvt_pk_bf16_f32
    return cv.u;
}
#else
__device__ __forceinline__ unsigned int pack2bf(float lo, float hi) {
    union { v2bf v; unsigned int u; } cv;
    cv.v = v2bf{(__bf16)lo, (__bf16)hi};               // pairwise fptrunc
    return cv.u;
}
#endif

// A-fragment (16x32, 16-bit): VGPR i of lane L holds K pair starting at:
__device__ __forceinline__ int kA(int i, int half) { return 2 * i + 8 * half + ((i >= 4) ? 8 : 0); }
// B-fragment (32x16, 16-bit): VGPR i of lane L holds K pair starting at:
__device__ __forceinline__ int kB(int i, int half) { return 2 * i + 8 * half + ((i >= 4) ? 8 : 0); }

__device__ __forceinline__ v8f wmma_bf16(const Frag16& a, const Frag16& b, v8f c) {
    return __builtin_amdgcn_wmma_f32_16x16x32_bf16(
        false, a.v, false, b.v, (short)0, c, false, false);
}

// ---------------------------------------------------------------------------
// Kernel 0: elementwise fp32 -> bf16 pair conversion (one pass; operands are
// reused many times by the GEMMs, so convert once, not per tile).
// ---------------------------------------------------------------------------
__global__ __launch_bounds__(256) void cvt_bf16_kernel(
    const float* __restrict__ src, unsigned int* __restrict__ dst, int npairs)
{
    const int idx = blockIdx.x * 256 + threadIdx.x;
    if (idx < npairs) {
        const float2 v = ((const float2*)src)[idx];
        dst[idx] = pack2bf(v.x, v.y);
    }
}

// ---------------------------------------------------------------------------
// Kernel 1: qkv = X(8192x768) @ W^T(768x2304) + bias, bf16 operands, f32 accum.
// Each wave owns a 16(M) x 64(N) tile: A fragment loaded once per K-step and
// reused by 4 WMMAs. Row m = s*B + b (native x layout AND output layout).
// Writes fp32 q/k/v to d_out regions 0/1/2 and bf16 staging for attention.
// ---------------------------------------------------------------------------
__global__ __launch_bounds__(256) void qkv_proj_kernel(
    const unsigned short* __restrict__ xb,     // (8192, 768) bf16
    const unsigned short* __restrict__ wb,     // (2304, 768) bf16
    const float* __restrict__ bias, float* __restrict__ dout,
    unsigned short* __restrict__ qb,   // (B*H, S, DH) bf16
    unsigned short* __restrict__ kbuf, // (B*H, S, DH) bf16
    unsigned short* __restrict__ vbt)  // (B*H, DH, S) bf16 (transposed)
{
    const int lane = threadIdx.x & 31;
    const int col  = lane & 15;
    const int half = lane >> 4;
    const int warp = __builtin_amdgcn_readfirstlane(threadIdx.x >> 5);
    const int wid  = blockIdx.x * 8 + warp;            // scalar
    const int tn   = wid % 36;                         // 2304/64
    const int tm   = wid / 36;                         // 8192/16
    const int m0   = tm * 16, n0 = tn * 64;
    const int seg  = tn / 12;                          // scalar: 0:q 1:k 2:v
    const int dbase = n0 - seg * D_;                   // scalar

    v8f acc[4] = {{}, {}, {}, {}};
    for (int k0 = 0; k0 < D_; k0 += 32) {
        Frag16 a;
#pragma unroll
        for (int i = 0; i < 8; ++i)
            a.d[i] = *(const unsigned int*)(xb + (size_t)(m0 + col) * D_ + k0 + kA(i, half));
#pragma unroll
        for (int j = 0; j < 4; ++j) {
            Frag16 bfr;
#pragma unroll
            for (int i = 0; i < 8; ++i)
                bfr.d[i] = *(const unsigned int*)(wb + (size_t)(n0 + j * 16 + col) * D_ + k0 + kB(i, half));
            acc[j] = wmma_bf16(a, bfr, acc[j]);
        }
    }

#pragma unroll
    for (int j = 0; j < 4; ++j) {
        const int d   = dbase + j * 16 + col;          // 0..767
        const float bval = bias[seg * D_ + d];
        const int h   = d >> 6;
        const int cd  = d & 63;
#pragma unroll
        for (int i = 0; i < 8; ++i) {
            const int m = m0 + i + 8 * half;
            const float v = acc[j][i] + bval;
            dout[(size_t)seg * QSZ_ + (size_t)m * D_ + d] = v;
            const int s = m >> 3, b = m & 7;
            const int bh = b * H_ + h;
            const unsigned short hv = f2bfu(v);
            if (seg == 0)      qb[((size_t)bh * S_ + s) * DH_ + cd] = hv;     // uniform branch
            else if (seg == 1) kbuf[((size_t)bh * S_ + s) * DH_ + cd] = hv;
            else               vbt[((size_t)bh * DH_ + cd) * S_ + s] = hv;
        }
    }
}

// ---------------------------------------------------------------------------
// Kernel 2: flash attention per (b,h). One 16-query tile per wave, 32-key blocks.
// ---------------------------------------------------------------------------
__global__ __launch_bounds__(256) void attn_kernel(
    const unsigned short* __restrict__ qb,
    const unsigned short* __restrict__ kbuf,
    const unsigned short* __restrict__ vbt,
    unsigned short* __restrict__ mixb)   // (S*B, D) bf16, row m = s*B+b
{
    __shared__ unsigned short pb[8][16 * 32];   // per-wave P staging (bf16)

    const int w    = __builtin_amdgcn_readfirstlane(threadIdx.x >> 5);
    const int lane = threadIdx.x & 31;
    const int col  = lane & 15;
    const int half = lane >> 4;
    const int wid  = blockIdx.x * 8 + w;        // scalar
    const int bh   = wid >> 6;                  // 64 query tiles per (b,h)
    const int q0   = (wid & 63) << 4;
    const int b    = bh / H_;
    const int h    = bh % H_;

    // Q fragments (kept in registers for the whole key loop)
    Frag16 aq[2];
#pragma unroll
    for (int c0 = 0; c0 < 2; ++c0)
#pragma unroll
        for (int i = 0; i < 8; ++i)
            aq[c0].d[i] = *(const unsigned int*)
                (qb + ((size_t)bh * S_ + q0 + col) * DH_ + c0 * 32 + kA(i, half));

    float mrun[8], lrun[8];
    v8f oacc[4] = {{}, {}, {}, {}};
#pragma unroll
    for (int i = 0; i < 8; ++i) { mrun[i] = -1.0e30f; lrun[i] = 0.0f; }

    for (int kb0 = 0; kb0 < S_; kb0 += 32) {
        if (kb0 + 32 < S_) {
            __builtin_prefetch(kbuf + ((size_t)bh * S_ + kb0 + 32) * DH_, 0, 1);
            __builtin_prefetch(vbt + (size_t)bh * DH_ * S_ + kb0 + 32, 0, 1);
        }
        float st[2][8];
#pragma unroll
        for (int t = 0; t < 2; ++t) {
            const int key0 = kb0 + t * 16;
            v8f sa = {};
#pragma unroll
            for (int c0 = 0; c0 < 2; ++c0) {
                Frag16 kf;
#pragma unroll
                for (int i = 0; i < 8; ++i)
                    kf.d[i] = *(const unsigned int*)
                        (kbuf + ((size_t)bh * S_ + key0 + col) * DH_ + c0 * 32 + kB(i, half));
                sa = wmma_bf16(aq[c0], kf, sa);
            }
#pragma unroll
            for (int i = 0; i < 8; ++i) {
                const int q  = q0 + i + 8 * half;
                const int kk = key0 + col;
                float sv = sa[i] * 0.125f;                       // 1/sqrt(64)
                if (kk >= TOKENS_ && (q == 0 || q >= TOKENS_)) sv = -10000.0f;
                st[t][i] = sv;
            }
        }
        // online softmax update (row lives in a 16-lane group of one VGPR)
#pragma unroll
        for (int i = 0; i < 8; ++i) {
            float mt = fmaxf(st[0][i], st[1][i]);
            mt = fmaxf(mt, __shfl_xor(mt, 1, 32));
            mt = fmaxf(mt, __shfl_xor(mt, 2, 32));
            mt = fmaxf(mt, __shfl_xor(mt, 4, 32));
            mt = fmaxf(mt, __shfl_xor(mt, 8, 32));
            const float nm   = fmaxf(mrun[i], mt);
            const float corr = __expf(mrun[i] - nm);
            mrun[i] = nm;
            const float p0 = __expf(st[0][i] - nm);
            const float p1 = __expf(st[1][i] - nm);
            st[0][i] = p0; st[1][i] = p1;
            float rs = p0 + p1;
            rs += __shfl_xor(rs, 1, 32);
            rs += __shfl_xor(rs, 2, 32);
            rs += __shfl_xor(rs, 4, 32);
            rs += __shfl_xor(rs, 8, 32);
            lrun[i] = lrun[i] * corr + rs;
#pragma unroll
            for (int n = 0; n < 4; ++n) oacc[n][i] *= corr;
        }
        // stage P (C-layout -> row-major 16x32 bf16 in LDS), wave-private
#pragma unroll
        for (int t = 0; t < 2; ++t)
#pragma unroll
            for (int i = 0; i < 8; ++i)
                pb[w][(i + 8 * half) * 32 + t * 16 + col] = f2bfu(st[t][i]);
        asm volatile("s_wait_dscnt 0" ::: "memory");
        Frag16 ap;
#pragma unroll
        for (int i = 0; i < 8; ++i)
            ap.d[i] = *(const unsigned int*)&pb[w][col * 32 + kA(i, half)];
        // O += P @ V  (V transposed: pairs contiguous along key)
#pragma unroll
        for (int n = 0; n < 4; ++n) {
            Frag16 bv;
#pragma unroll
            for (int i = 0; i < 8; ++i)
                bv.d[i] = *(const unsigned int*)
                    (vbt + ((size_t)bh * DH_ + n * 16 + col) * S_ + kb0 + kB(i, half));
            oacc[n] = wmma_bf16(ap, bv, oacc[n]);
        }
    }

    // normalize and store mix (bf16) in (m = s*B+b, D) layout
#pragma unroll
    for (int i = 0; i < 8; ++i) {
        const float inv = 1.0f / lrun[i];
        const int s = q0 + i + 8 * half;
        const int m = s * B_ + b;
#pragma unroll
        for (int n = 0; n < 4; ++n)
            mixb[(size_t)m * D_ + h * DH_ + n * 16 + col] = f2bfu(oacc[n][i] * inv);
    }
}

// ---------------------------------------------------------------------------
// Kernel 3: out = mix(8192x768) @ out_w^T(768x768) + out_b  -> d_out region 3
// 16(M) x 64(N) tile per wave, all-bf16 operands.
// ---------------------------------------------------------------------------
__global__ __launch_bounds__(256) void out_proj_kernel(
    const unsigned short* __restrict__ mixb, const unsigned short* __restrict__ wb,
    const float* __restrict__ bias, float* __restrict__ dout)
{
    const int lane = threadIdx.x & 31;
    const int col  = lane & 15;
    const int half = lane >> 4;
    const int warp = __builtin_amdgcn_readfirstlane(threadIdx.x >> 5);
    const int wid  = blockIdx.x * 8 + warp;    // scalar
    const int tn   = wid % 12;                 // 768/64
    const int tm   = wid / 12;                 // 8192/16
    const int m0   = tm * 16, n0 = tn * 64;

    v8f acc[4] = {{}, {}, {}, {}};
    for (int k0 = 0; k0 < D_; k0 += 32) {
        Frag16 a;
#pragma unroll
        for (int i = 0; i < 8; ++i)
            a.d[i] = *(const unsigned int*)(mixb + (size_t)(m0 + col) * D_ + k0 + kA(i, half));
#pragma unroll
        for (int j = 0; j < 4; ++j) {
            Frag16 bfr;
#pragma unroll
            for (int i = 0; i < 8; ++i)
                bfr.d[i] = *(const unsigned int*)(wb + (size_t)(n0 + j * 16 + col) * D_ + k0 + kB(i, half));
            acc[j] = wmma_bf16(a, bfr, acc[j]);
        }
    }
#pragma unroll
    for (int j = 0; j < 4; ++j) {
        const float bval = bias[n0 + j * 16 + col];
#pragma unroll
        for (int i = 0; i < 8; ++i) {
            const int m = m0 + i + 8 * half;
            dout[(size_t)3 * QSZ_ + (size_t)m * D_ + (n0 + j * 16 + col)] = acc[j][i] + bval;
        }
    }
}

// ---------------------------------------------------------------------------
extern "C" void kernel_launch(void* const* d_in, const int* in_sizes, int n_in,
                              void* d_out, int out_size, void* d_ws, size_t ws_size,
                              hipStream_t stream) {
    const float* x    = (const float*)d_in[0];
    const float* wqkv = (const float*)d_in[1];
    const float* bqkv = (const float*)d_in[2];
    const float* wout = (const float*)d_in[3];
    const float* bout = (const float*)d_in[4];
    float* dout = (float*)d_out;

    // workspace layout (bf16 halves):
    //   qb | kb | vbt | mixb (each NQ) | xb (NQ) | wqkvb (3*D*D) | woutb (D*D)
    const size_t NQ = (size_t)B_ * H_ * S_ * DH_;   // == S*B*D == 6291456
    unsigned short* qb     = (unsigned short*)d_ws;
    unsigned short* kbuf   = qb + NQ;
    unsigned short* vbt    = kbuf + NQ;
    unsigned short* mixb   = vbt + NQ;
    unsigned short* xb     = mixb + NQ;
    unsigned short* wqkvb  = xb + NQ;
    unsigned short* woutb  = wqkvb + (size_t)3 * D_ * D_;

    // one-pass fp32 -> bf16 conversions (operands reused 36x/512x by GEMMs)
    const int xp = QSZ_ / 2;                 // 3145728 pairs
    const int wp = 3 * D_ * D_ / 2;          // 884736 pairs
    const int op = D_ * D_ / 2;              // 294912 pairs
    cvt_bf16_kernel<<<(xp + 255) / 256, 256, 0, stream>>>(x, (unsigned int*)xb, xp);
    cvt_bf16_kernel<<<(wp + 255) / 256, 256, 0, stream>>>(wqkv, (unsigned int*)wqkvb, wp);
    cvt_bf16_kernel<<<(op + 255) / 256, 256, 0, stream>>>(wout, (unsigned int*)woutb, op);

    // 512 m-tiles * 36 n-tiles / 8 waves per 256-thread block
    qkv_proj_kernel<<<2304, 256, 0, stream>>>(xb, wqkvb, bqkv, dout, qb, kbuf, vbt);
    // 96 (b,h) * 64 query tiles / 8 waves per block
    attn_kernel<<<768, 256, 0, stream>>>(qb, kbuf, vbt, mixb);
    // 512 m-tiles * 12 n-tiles / 8 waves per block
    out_proj_kernel<<<768, 256, 0, stream>>>(mixb, woutb, bout, dout);
}
